// DeterministicEgnnPolicy_82188494176620
// MI455X (gfx1250) — matrix operand order repaired and checked by reading, not compiled, not run on hardware
//
#include <hip/hip_runtime.h>

// ---------------------------------------------------------------------------
// EGNN policy, fully fused: one workgroup (8 wave32s) per batch-group of 32
// agents. All state (h, loc, vel, per-layer weights) lives in LDS; every
// GEMM runs on v_wmma_f32_16x16x32_f16 with f32 accumulation.
// Key reduction: eh @ We1 == A[i] + B[j] + radial*w128 + ea*w129 with
// A = h @ We1[0:64], B = h @ We1[64:128] computed per-node.
// Diagonal (i==j) edges are included (diff==0 -> trans==0 automatically)
// and subtracted out of the magg reduction.
// Elementwise stages use b128/b64 LDS vector ops; silu uses v_rcp_f32.
// ---------------------------------------------------------------------------

typedef __attribute__((ext_vector_type(16))) _Float16 v16h;
typedef __attribute__((ext_vector_type(8)))  float    v8f;
typedef __attribute__((ext_vector_type(4)))  float    f4;
typedef __attribute__((ext_vector_type(4)))  _Float16 h4;

#define NAG      32          // agents per group
#define NGROUPS  1024
#define HDIM     64
#define NLAYERS  4
#define INV_DIM  16
#define OBS_C    20          // INV + loc(2) + vel(2)
#define NTHREADS 256
#define NWAVES   8
#define CHUNK    128         // edge rows per chunk (4 source nodes, 8 M-tiles)
#define NCHUNK   8           // 1024 / 128
#define STH      72          // f16 row stride: 144 B = 9*16, b128-aligned rows
#define STC      144         // f16 row stride for concat [h | magg] buffers
#define STF      68          // f32 row stride: 272 B = 17*16, b128-aligned rows

struct Smem {
  alignas(16) float    h32[NAG][STF];      // master h, fp32
  alignas(16) _Float16 catf[NAG][STC];     // [ h_f16 | magg_f16 ]
  alignas(16) float    invs[NAG][17];
  float    locx[NAG], locy[NAG], l0x[NAG], l0y[NAG], velx[NAG], vely[NAG];
  alignas(16) float    A32[NAG][STF], B32[NAG][STF];
  alignas(16) _Float16 uf[NAG][STH];       // silu(h@Wv1+bv1), later silu(cat@Wn1+bn1)
  float    phiv[NAG], aggx[NAG], aggy[NAG];
  alignas(16) _Float16 bufA[CHUNK][STH];   // edge stage buffers (ping-pong)
  alignas(16) _Float16 bufB[CHUNK][STH];
  alignas(16) _Float16 We1aT[HDIM][STH], We1bT[HDIM][STH], We2T[HDIM][STH];
  alignas(16) _Float16 Wc1T[HDIM][STH],  Wv1T[HDIM][STH],  Wn2T[HDIM][STH];
  alignas(16) _Float16 Wn1T[HDIM][STC];    // 64 x 128 transposed
  alignas(16) float    w128[HDIM], w129[HDIM];
  alignas(16) float    be1v[HDIM], be2v[HDIM], bc1v[HDIM];
  alignas(16) float    bv1v[HDIM], bn1v[HDIM], bn2v[HDIM];
  alignas(16) float    wc2v[HDIM], wv2v[HDIM];
  float    bc2s, bv2s;
};

// silu via hardware reciprocal (v_rcp_f32), avoids IEEE division sequence.
__device__ __forceinline__ float silu_f(float x) {
  return x * __builtin_amdgcn_rcpf(1.0f + __expf(-x));
}

// A fragment (16x32 f16, MxK) from row-major LDS. ISA layout:
// lane m=lane&15, g=lane>>4; VGPR0-3: K = g*8 + {0..7}; VGPR4-7: K = 16+g*8+{0..7}
__device__ __forceinline__ v16h load_a_frag(const _Float16* p, int stride, int m0, int k0) {
  const int lane = threadIdx.x & 31;
  const int m = lane & 15, g = lane >> 4;
  const _Float16* row = p + (m0 + m) * stride + k0;
  v16h a;
#pragma unroll
  for (int v = 0; v < 8; ++v) {
    const int kk = (v < 4) ? (g * 8 + 2 * v) : (16 + g * 8 + 2 * (v - 4));
    a[2 * v]     = row[kk];
    a[2 * v + 1] = row[kk + 1];
  }
  return a;
}

// B fragment (32x16 f16, KxN) from TRANSPOSED LDS weights WT[n][k] so that
// the (K, K+1) pair per VGPR is contiguous. ISA layout: lane n=lane&15,
// g=lane>>4; VGPR0-3: K = g*8 + {0..7}; VGPR4-7: K = 16+g*8+{0..7}.
__device__ __forceinline__ v16h load_b_frag(const _Float16* pT, int stride, int n0, int k0) {
  const int lane = threadIdx.x & 31;
  const int n = lane & 15, g = lane >> 4;
  const _Float16* row = pT + (n0 + n) * stride + k0;
  v16h b;
#pragma unroll
  for (int v = 0; v < 8; ++v) {
    const int kk = (v < 4) ? (g * 8 + 2 * v) : (16 + g * 8 + 2 * (v - 4));
    b[2 * v]     = row[kk];
    b[2 * v + 1] = row[kk + 1];
  }
  return b;
}

__device__ __forceinline__ v8f wmma_acc(v16h a, v16h b, v8f c) {
  return __builtin_amdgcn_wmma_f32_16x16x32_f16(false, a, false, b,
                                                (short)0, c, false, false);
}

// K=64 tile with pre-loaded A fragments.
__device__ __forceinline__ v8f wmma_k64(v16h a0, v16h a1,
                                        const _Float16* BT, int bs, int n0) {
  v8f acc = {};
  acc = wmma_acc(a0, load_b_frag(BT, bs, n0, 0),  acc);
  acc = wmma_acc(a1, load_b_frag(BT, bs, n0, 32), acc);
  return acc;
}

__global__ __launch_bounds__(NTHREADS)
void egnn_fused_kernel(const float* __restrict__ obs,
                       const float* __restrict__ scale, const float* __restrict__ mean,
                       const float* __restrict__ embW,  const float* __restrict__ embB,
                       const float* __restrict__ We1,   const float* __restrict__ be1,
                       const float* __restrict__ We2,   const float* __restrict__ be2,
                       const float* __restrict__ Wc1,   const float* __restrict__ bc1,
                       const float* __restrict__ Wc2,   const float* __restrict__ bc2,
                       const float* __restrict__ Wv1,   const float* __restrict__ bv1,
                       const float* __restrict__ Wv2,   const float* __restrict__ bv2,
                       const float* __restrict__ Wn1,   const float* __restrict__ bn1,
                       const float* __restrict__ Wn2,   const float* __restrict__ bn2,
                       float* __restrict__ out) {
  __shared__ Smem sm;
  const int tid   = threadIdx.x;
  const int lane  = tid & 31;
  // wave index is uniform per wave: force it into an SGPR so all tile-index
  // math and branches around WMMA are scalar (no EXEC manipulation).
  const int wave  = __builtin_amdgcn_readfirstlane(tid >> 5);
  const int g     = blockIdx.x;
  const int node0 = g * NAG;

  // C/D fragment element coordinates for this lane (16x16 f32 tile):
  // element r sits at (row = r + 8*cg, col = cn) within the tile.
  const int cg = lane >> 4;            // 0 or 1
  const int cn = lane & 15;            // column within tile

  // ---- load obs for this group ----
  for (int idx = tid; idx < NAG * OBS_C; idx += NTHREADS) {
    const int i = idx / OBS_C, c = idx % OBS_C;
    const float v = obs[(node0 + i) * OBS_C + c];
    if (c < INV_DIM)      sm.invs[i][c] = v;
    else if (c == 16) { sm.locx[i] = v; sm.l0x[i] = v; }
    else if (c == 17) { sm.locy[i] = v; sm.l0y[i] = v; }
    else if (c == 18)   sm.velx[i] = v;
    else                sm.vely[i] = v;
  }
  __syncthreads();

  // ---- embedding: h = inv @ embW + embB (fp32 VALU, tiny) ----
  for (int idx = tid; idx < NAG * HDIM; idx += NTHREADS) {
    const int i = idx >> 6, c = idx & 63;
    float acc = embB[c];
#pragma unroll
    for (int k = 0; k < INV_DIM; ++k) acc += sm.invs[i][k] * embW[k * HDIM + c];
    sm.h32[i][c] = acc;
  }
  __syncthreads();

  for (int l = 0; l < NLAYERS; ++l) {
    // ---- stage layer weights into LDS (f16, transposed) ----
    const float* gWe1 = We1 + l * 130 * HDIM;
    const float* gWe2 = We2 + l * HDIM * HDIM;
    const float* gWc1 = Wc1 + l * HDIM * HDIM;
    const float* gWv1 = Wv1 + l * HDIM * HDIM;
    const float* gWn1 = Wn1 + l * 2 * HDIM * HDIM;
    const float* gWn2 = Wn2 + l * HDIM * HDIM;
    for (int idx = tid; idx < HDIM * HDIM; idx += NTHREADS) {
      const int k = idx >> 6, n = idx & 63;
      sm.We1aT[n][k]     = (_Float16)gWe1[k * HDIM + n];
      sm.We1bT[n][k]     = (_Float16)gWe1[(64 + k) * HDIM + n];
      sm.We2T[n][k]      = (_Float16)gWe2[k * HDIM + n];
      sm.Wc1T[n][k]      = (_Float16)gWc1[k * HDIM + n];
      sm.Wv1T[n][k]      = (_Float16)gWv1[k * HDIM + n];
      sm.Wn2T[n][k]      = (_Float16)gWn2[k * HDIM + n];
      sm.Wn1T[n][k]      = (_Float16)gWn1[k * HDIM + n];
      sm.Wn1T[n][64 + k] = (_Float16)gWn1[(64 + k) * HDIM + n];
    }
    if (tid < HDIM) {
      sm.w128[tid] = gWe1[128 * HDIM + tid];
      sm.w129[tid] = gWe1[129 * HDIM + tid];
      sm.be1v[tid] = be1[l * HDIM + tid];
      sm.be2v[tid] = be2[l * HDIM + tid];
      sm.bc1v[tid] = bc1[l * HDIM + tid];
      sm.bv1v[tid] = bv1[l * HDIM + tid];
      sm.bn1v[tid] = bn1[l * HDIM + tid];
      sm.bn2v[tid] = bn2[l * HDIM + tid];
      sm.wc2v[tid] = Wc2[l * HDIM + tid];
      sm.wv2v[tid] = Wv2[l * HDIM + tid];
    }
    if (tid == 0) { sm.bc2s = bc2[l]; sm.bv2s = bv2[l]; }
    // h fp32 -> f16 into concat buffer cols [0,64)
    for (int idx = tid; idx < NAG * HDIM; idx += NTHREADS) {
      const int i = idx >> 6, c = idx & 63;
      sm.catf[i][c] = (_Float16)sm.h32[i][c];
    }
    __syncthreads();

    // ---- node-level WMMA: A = h@We1a, B = h@We1b, u = silu(h@Wv1+bv1) ----
    // 2 M-tiles x 4 N-tiles: exactly one tile per wave per matrix.
    {
      const int mt = (wave >> 2) & 1, nt = wave & 3;
      const int m0 = mt * 16, n0 = nt * 16;
      // A-frags of h are shared by all three matrices.
      const v16h a0 = load_a_frag(&sm.catf[0][0], STC, m0, 0);
      const v16h a1 = load_a_frag(&sm.catf[0][0], STC, m0, 32);

      const v8f accA = wmma_k64(a0, a1, &sm.We1aT[0][0], STH, n0);
      const v8f accB = wmma_k64(a0, a1, &sm.We1bT[0][0], STH, n0);
      const v8f accU = wmma_k64(a0, a1, &sm.Wv1T[0][0],  STH, n0);
      const int col = n0 + cn;
#pragma unroll
      for (int r = 0; r < 8; ++r) {
        const int row = m0 + r + 8 * cg;
        sm.A32[row][col] = accA[r];
        sm.B32[row][col] = accB[r];
        sm.uf[row][col]  = (_Float16)silu_f(accU[r] + sm.bv1v[col]);
      }
    }
    __syncthreads();

    // phi_v = u @ Wv2 + bv2  (b64 f16 loads + vector MACs)
    if (tid < NAG) {
      const h4* uq = (const h4*)&sm.uf[tid][0];
      const f4* wq = (const f4*)&sm.wv2v[0];
      f4 acc4 = {};
#pragma unroll 4
      for (int c4 = 0; c4 < 16; ++c4) {
        const h4 u = uq[c4];
        const f4 w = wq[c4];
        acc4 += (f4){(float)u[0], (float)u[1], (float)u[2], (float)u[3]} * w;
      }
      sm.phiv[tid] = sm.bv2s + acc4[0] + acc4[1] + acc4[2] + acc4[3];
    }

    // ---- edge pipeline: 1024 edge rows in chunks of 128 (4 source nodes) ----
    for (int ch = 0; ch < NCHUNK; ++ch) {
      const int r0 = ch * CHUNK;
      const int i  = (r0 + tid) >> 5;   // source node of this thread's edge
      const int j  = tid & 31;          // dest node
      float mydx, mydy;
      {
        const float dx = sm.locx[i] - sm.locx[j];
        const float dy = sm.locy[i] - sm.locy[j];
        const float rad = dx * dx + dy * dy;
        const float rinv = __builtin_amdgcn_rcpf(__builtin_amdgcn_sqrtf(rad) + 1.0f);
        mydx = dx * rinv; mydy = dy * rinv;
        const float ex = sm.l0x[i] - sm.l0x[j];
        const float ey = sm.l0y[i] - sm.l0y[j];
        const float ea = ex * ex + ey * ey;
        // stage 1: m1 = silu(A[i] + B[j] + rad*w128 + ea*w129 + be1),
        // vectorized: b128 LDS loads, b64 f16 stores.
        const f4* Ai = (const f4*)&sm.A32[i][0];
        const f4* Bj = (const f4*)&sm.B32[j][0];
        const f4* W1 = (const f4*)&sm.w128[0];
        const f4* W2 = (const f4*)&sm.w129[0];
        const f4* Bb = (const f4*)&sm.be1v[0];
#pragma unroll 4
        for (int c4 = 0; c4 < 16; ++c4) {
          const f4 val = Ai[c4] + Bj[c4] + rad * W1[c4] + ea * W2[c4] + Bb[c4];
          h4 o;
#pragma unroll
          for (int e = 0; e < 4; ++e) o[e] = (_Float16)silu_f(val[e]);
          *(h4*)&sm.bufA[tid][4 * c4] = o;
        }
      }
      __syncthreads();

      // stage 2 (WMMA): m = silu(m1 @ We2 + be2) -> bufB.
      // 8 M-tiles: mt == wave; A-frags hoisted across the 4 N-tiles.
      {
        const int m0 = wave * 16;
        const v16h a0 = load_a_frag(&sm.bufA[0][0], STH, m0, 0);
        const v16h a1 = load_a_frag(&sm.bufA[0][0], STH, m0, 32);
#pragma unroll
        for (int nt = 0; nt < 4; ++nt) {
          const v8f acc = wmma_k64(a0, a1, &sm.We2T[0][0], STH, nt * 16);
          const int col = nt * 16 + cn;
#pragma unroll
          for (int r = 0; r < 8; ++r)
            sm.bufB[m0 + r + 8 * cg][col] = (_Float16)silu_f(acc[r] + sm.be2v[col]);
        }
      }
      __syncthreads();

      // magg for this chunk's 4 source nodes: sum all 32 rows, subtract the
      // diagonal term (branch-free). Written f16 into concat cols [64,128).
      {
        const int nodeBase = r0 >> 5;           // ch*4
        const int ii = tid >> 6, c = tid & 63;  // 4 nodes x 64 cols = 256 jobs
        float s = 0.0f;
#pragma unroll 8
        for (int q = 0; q < NAG; ++q) s += (float)sm.bufB[ii * NAG + q][c];
        s -= (float)sm.bufB[ii * NAG + nodeBase + ii][c];
        sm.catf[nodeBase + ii][64 + c] = (_Float16)s;
      }

      // stage 3 (WMMA): q = silu(m @ Wc1 + bc1) -> bufA (m1 dead)
      {
        const int m0 = wave * 16;
        const v16h a0 = load_a_frag(&sm.bufB[0][0], STH, m0, 0);
        const v16h a1 = load_a_frag(&sm.bufB[0][0], STH, m0, 32);
#pragma unroll
        for (int nt = 0; nt < 4; ++nt) {
          const v8f acc = wmma_k64(a0, a1, &sm.Wc1T[0][0], STH, nt * 16);
          const int col = nt * 16 + cn;
#pragma unroll
          for (int r = 0; r < 8; ++r)
            sm.bufA[m0 + r + 8 * cg][col] = (_Float16)silu_f(acc[r] + sm.bc1v[col]);
        }
      }
      __syncthreads();

      // stage 4: s = q.wc2 + bc2 ; trans = diff*s ; wave-reduce over j
      {
        const h4* qq = (const h4*)&sm.bufA[tid][0];
        const f4* wq = (const f4*)&sm.wc2v[0];
        f4 acc4 = {};
#pragma unroll 4
        for (int c4 = 0; c4 < 16; ++c4) {
          const h4 u = qq[c4];
          const f4 w = wq[c4];
          acc4 += (f4){(float)u[0], (float)u[1], (float)u[2], (float)u[3]} * w;
        }
        const float s = sm.bc2s + acc4[0] + acc4[1] + acc4[2] + acc4[3];
        float tx = mydx * s, ty = mydy * s;   // diagonal: diff==0 -> 0
#pragma unroll
        for (int off = 16; off > 0; off >>= 1) {
          tx += __shfl_xor(tx, off, 32);
          ty += __shfl_xor(ty, off, 32);
        }
        if (lane == 0) { sm.aggx[i] = tx; sm.aggy[i] = ty; }
      }
      __syncthreads();
    }

    // ---- vel/loc update ----
    if (tid < NAG) {
      const float vx = sm.phiv[tid] * sm.velx[tid] + sm.aggx[tid] * (1.0f / 31.0f);
      const float vy = sm.phiv[tid] * sm.vely[tid] + sm.aggy[tid] * (1.0f / 31.0f);
      sm.velx[tid] = vx; sm.vely[tid] = vy;
      sm.locx[tid] += vx; sm.locy[tid] += vy;
    }
    __syncthreads();

    // ---- n1 = silu([h|magg] @ Wn1 + bn1) -> uf  (K = 128, one tile/wave) ----
    {
      const int mt = (wave >> 2) & 1, nt = wave & 3;
      const int m0 = mt * 16, n0 = nt * 16;
      v8f acc = {};
#pragma unroll
      for (int kt = 0; kt < 4; ++kt) {
        const v16h a = load_a_frag(&sm.catf[0][0], STC, m0, kt * 32);
        const v16h b = load_b_frag(&sm.Wn1T[0][0], STC, n0, kt * 32);
        acc = wmma_acc(a, b, acc);
      }
      const int col = n0 + cn;
#pragma unroll
      for (int r = 0; r < 8; ++r)
        sm.uf[m0 + r + 8 * cg][col] = (_Float16)silu_f(acc[r] + sm.bn1v[col]);
    }
    __syncthreads();

    // ---- h += n1 @ Wn2 + bn2 (one tile per wave) ----
    {
      const int mt = (wave >> 2) & 1, nt = wave & 3;
      const int m0 = mt * 16, n0 = nt * 16;
      const v16h a0 = load_a_frag(&sm.uf[0][0], STH, m0, 0);
      const v16h a1 = load_a_frag(&sm.uf[0][0], STH, m0, 32);
      const v8f acc = wmma_k64(a0, a1, &sm.Wn2T[0][0], STH, n0);
      const int col = n0 + cn;
#pragma unroll
      for (int r = 0; r < 8; ++r) {
        const int row = m0 + r + 8 * cg;
        sm.h32[row][col] += acc[r] + sm.bn2v[col];
      }
    }
    __syncthreads();
  }

  // ---- output: scale * vel + mean ----
  if (tid < NAG) {
    const int n = node0 + tid;
    out[2 * n]     = scale[0] * sm.velx[tid] + mean[0];
    out[2 * n + 1] = scale[1] * sm.vely[tid] + mean[1];
  }
}

extern "C" void kernel_launch(void* const* d_in, const int* in_sizes, int n_in,
                              void* d_out, int out_size, void* d_ws, size_t ws_size,
                              hipStream_t stream) {
  (void)in_sizes; (void)n_in; (void)d_ws; (void)ws_size; (void)out_size;
  const float* obs   = (const float*)d_in[0];
  // d_in[1] = rows, d_in[2] = cols : graph structure is block-dense, unused
  const float* scale = (const float*)d_in[3];
  const float* mean  = (const float*)d_in[4];
  const float* embW  = (const float*)d_in[5];
  const float* embB  = (const float*)d_in[6];
  const float* We1   = (const float*)d_in[7];
  const float* be1   = (const float*)d_in[8];
  const float* We2   = (const float*)d_in[9];
  const float* be2   = (const float*)d_in[10];
  const float* Wc1   = (const float*)d_in[11];
  const float* bc1   = (const float*)d_in[12];
  const float* Wc2   = (const float*)d_in[13];
  const float* bc2   = (const float*)d_in[14];
  const float* Wv1   = (const float*)d_in[15];
  const float* bv1   = (const float*)d_in[16];
  const float* Wv2   = (const float*)d_in[17];
  const float* bv2   = (const float*)d_in[18];
  const float* Wn1   = (const float*)d_in[19];
  const float* bn1   = (const float*)d_in[20];
  const float* Wn2   = (const float*)d_in[21];
  const float* bn2   = (const float*)d_in[22];

  egnn_fused_kernel<<<NGROUPS, NTHREADS, 0, stream>>>(
      obs, scale, mean, embW, embB, We1, be1, We2, be2, Wc1, bc1, Wc2, bc2,
      Wv1, bv1, Wv2, bv2, Wn1, bn1, Wn2, bn2, (float*)d_out);
}
